// Decoder_9311489098140
// MI455X (gfx1250) — matrix operand order repaired and checked by reading, not compiled, not run on hardware
//
#include <hip/hip_runtime.h>
#include <hip/hip_bf16.h>

typedef __attribute__((ext_vector_type(16))) _Float16 v16h;
typedef __attribute__((ext_vector_type(8)))  _Float16 v8h;
typedef __attribute__((ext_vector_type(2)))  __fp16   fp16x2;
typedef __attribute__((ext_vector_type(8)))  float    v8f;
typedef unsigned int uint;

#define WAVES1 4
#define WAVES2 8

static __device__ __forceinline__ void lds_fence() {
    asm volatile("s_wait_dscnt 0" ::: "memory");
}

// jax.nn.gelu (tanh approximation), rewritten as a sigmoid:
//   gelu(x) = x * sigmoid(1.5957691*(x + 0.044715 x^3))
//           = x * rcp(1 + exp2(c1*x + c2*x^3))
// branchless: one v_exp_f32 + one v_rcp_f32 + FMAs; saturates correctly.
static __device__ __forceinline__ float gelu_f(float x) {
    const float c1 = -2.302158167f;            // -1.5957691216 * log2(e)
    const float c2 = -0.102946005f;            // c1 * 0.044715
    float u = x * (c1 + c2 * x * x);
    float e = __builtin_amdgcn_exp2f(u);
    return x * __builtin_amdgcn_rcpf(1.0f + e);
}

// Load a 16x32 f16 A-fragment from an LDS tile stored row-major [16][64] halfs.
// Per ISA: lane<16 -> M=lane, halves 0..7 = K 0..7, halves 8..15 = K 16..23;
//          lane>=16 -> same M, K offset +8.
static __device__ __forceinline__ v16h ldA(const _Float16* base, int row, int khalf, int kb) {
    union { v16h v; v8h h[2]; } u;
    const _Float16* p = base + row * 64 + kb * 32 + khalf * 8;
    u.h[0] = *(const v8h*)p;
    u.h[1] = *(const v8h*)(p + 16);
    return u.v;
}

static __device__ __forceinline__ v8f wmma_f16(v16h a, v16h b, v8f c) {
    return __builtin_amdgcn_wmma_f32_16x16x32_f16(false, a, false, b, (short)0, c, false, false);
}

// ---------------- Kernel 0: compute pe(16x64) and pack B-fragments ----------------
// B-fragment layout (32x16 f16): lane l holds column n = l&15;
// half index H: k = (H&7) + 8*(l>>4) + 16*(H>>3)  (matches ldA A-fragment order).
static __device__ __forceinline__ void packB(_Float16* dst, const float* W, int K, int N,
                                             int ntiles, int tid, int nthreads) {
    int total = ntiles * 2 * 32 * 16;
    for (int i = tid; i < total; i += nthreads) {
        int h = i & 15;
        int lane = (i >> 4) & 31;
        int kb = (i >> 9) & 1;
        int nt = i >> 10;
        int n = nt * 16 + (lane & 15);
        int k = kb * 32 + (h & 7) + 8 * (lane >> 4) + 16 * (h >> 3);
        float v = (n < N && k < K) ? W[k * N + n] : 0.0f;
        dst[i] = (_Float16)v;
    }
}

__global__ __launch_bounds__(256) void k0_pack(
    const float* sp_w1, const float* sp_w2,
    const float* pe_w1, const float* pe_b1, const float* pe_w2, const float* pe_b2,
    const float* dec_w1, const float* dec_w2,
    float* ws_pe, _Float16* f_w1, _Float16* f_w2, _Float16* f_sw1, _Float16* f_sw2) {
    int tid = threadIdx.x;
    if (tid < 16) {
        int p = tid;
        float h[40];
        #pragma unroll
        for (int j = 0; j < 40; ++j) h[j] = gelu_f(pe_w1[p * 40 + j] + pe_b1[j]);
        for (int d = 0; d < 64; ++d) {
            float s = pe_b2[d];
            #pragma unroll
            for (int j = 0; j < 40; ++j) s += h[j] * pe_w2[j * 64 + d];
            ws_pe[p * 64 + d] = s;
        }
    }
    packB(f_w1,  dec_w1, 64, 64, 4, tid, blockDim.x);
    packB(f_w2,  dec_w2, 64, 64, 4, tid, blockDim.x);
    packB(f_sw1, sp_w1,  64, 40, 3, tid, blockDim.x);   // pad N 40->48
    packB(f_sw2, sp_w2,  40, 16, 1, tid, blockDim.x);   // pad K 40->64
}

// ---------------- Kernel 1: size-prediction MLP + argmax (WMMA, M=16 batch rows) ----
__global__ __launch_bounds__(128) void k1_sp(
    const float* __restrict__ z, const float* __restrict__ spb1, const float* __restrict__ spb2,
    const _Float16* __restrict__ w1f, const _Float16* __restrict__ w2f,
    int* __restrict__ nvec, float* __restrict__ outNpred, int B) {
    __shared__ __attribute__((aligned(16))) _Float16 zA[WAVES1][1024];
    __shared__ __attribute__((aligned(16))) _Float16 yB[WAVES1][1024];
    __shared__ __attribute__((aligned(16))) float    np[WAVES1][256];

    int w = threadIdx.x >> 5, lane = threadIdx.x & 31;
    int t = blockIdx.x * WAVES1 + w;
    if (t * 16 >= B) return;
    int t16 = t * 16;

    _Float16* Z = zA[w];
    // Stage 16 batch rows of z as f16, row-major [16][64]
    #pragma unroll
    for (int r = 0; r < 16; ++r) {
        float2 zz = ((const float2*)(z + (size_t)(t16 + r) * 64))[lane];
        *(fp16x2*)&Z[r * 64 + 2 * lane] = __builtin_amdgcn_cvt_pkrtz(zz.x, zz.y);
    }
    // Zero pad columns 48..63 of the intermediate tile
    {
        uint* yu = (uint*)&yB[w][(lane & 15) * 64 + 48 + (lane >> 4) * 8];
        yu[0] = 0; yu[1] = 0; yu[2] = 0; yu[3] = 0;
    }
    lds_fence();

    int col = lane & 15, khalf = lane >> 4;
    v16h a0 = ldA(Z, col, khalf, 0);
    v16h a1 = ldA(Z, col, khalf, 1);

    const v16h* W1 = (const v16h*)w1f;
    for (int nt = 0; nt < 3; ++nt) {
        int n = nt * 16 + col;
        float bb = (n < 40) ? spb1[n] : 0.0f;
        v8f c = {bb, bb, bb, bb, bb, bb, bb, bb};
        c = wmma_f16(a0, W1[(nt * 2 + 0) * 32 + lane], c);
        c = wmma_f16(a1, W1[(nt * 2 + 1) * 32 + lane], c);
        #pragma unroll
        for (int v = 0; v < 8; ++v)
            yB[w][(v + 8 * khalf) * 64 + n] = (_Float16)gelu_f(c[v]);
    }
    lds_fence();

    a0 = ldA(yB[w], col, khalf, 0);
    a1 = ldA(yB[w], col, khalf, 1);
    const v16h* W2 = (const v16h*)w2f;
    float bb2 = spb2[col];
    v8f c = {bb2, bb2, bb2, bb2, bb2, bb2, bb2, bb2};
    c = wmma_f16(a0, W2[lane], c);
    c = wmma_f16(a1, W2[32 + lane], c);

    #pragma unroll
    for (int v = 0; v < 8; ++v) {
        int m = v + 8 * khalf;
        outNpred[(size_t)(t16 + m) * 16 + col] = c[v];
        np[w][m * 16 + col] = c[v];
    }
    lds_fence();

    if (lane < 16) {
        const float* row = &np[w][lane * 16];
        float best = row[0]; int bi = 0;
        #pragma unroll
        for (int j = 1; j < 16; ++j) { float vv = row[j]; if (vv > best) { best = vv; bi = j; } }
        nvec[t16 + lane] = bi;
    }
}

// ---------------- Kernel 2: fused decoder MLP, one wave per batch element -----------
__global__ __launch_bounds__(256) void k2_dec(
    const float* __restrict__ z, const float* __restrict__ db1, const float* __restrict__ db2,
    const float* __restrict__ pe_g, const _Float16* __restrict__ w1f, const _Float16* __restrict__ w2f,
    const int* __restrict__ nvec,
    float* __restrict__ outX, float* __restrict__ outMask, float* __restrict__ outBatch, int B) {
    __shared__ __attribute__((aligned(16))) float    pe[1024];
    __shared__ __attribute__((aligned(16))) _Float16 zp[WAVES2][1024];
    __shared__ __attribute__((aligned(16))) _Float16 yb[WAVES2][1024];

    int tid = threadIdx.x;
    ((float4*)pe)[tid] = ((const float4*)pe_g)[tid];   // 256 threads * 16B = 4KB
    __syncthreads();

    int w = tid >> 5, lane = tid & 31;
    int b = blockIdx.x * WAVES2 + w;
    if (b >= B) return;

    float2 zz = ((const float2*)(z + (size_t)b * 64))[lane];
    _Float16* Z = zp[w];
    // zp[p][k] = z[k] * pe[p][k], f16 row-major [16][64]
    #pragma unroll
    for (int p = 0; p < 16; ++p) {
        float2 pp = *(const float2*)&pe[p * 64 + 2 * lane];
        *(fp16x2*)&Z[p * 64 + 2 * lane] = __builtin_amdgcn_cvt_pkrtz(zz.x * pp.x, zz.y * pp.y);
    }
    lds_fence();

    int col = lane & 15, khalf = lane >> 4;
    v16h a0 = ldA(Z, col, khalf, 0);
    v16h a1 = ldA(Z, col, khalf, 1);

    const v16h* W1 = (const v16h*)w1f;
    _Float16* Y = yb[w];
    for (int nt = 0; nt < 4; ++nt) {
        float bb = db1[nt * 16 + col];
        v8f c = {bb, bb, bb, bb, bb, bb, bb, bb};
        c = wmma_f16(a0, W1[(nt * 2 + 0) * 32 + lane], c);
        c = wmma_f16(a1, W1[(nt * 2 + 1) * 32 + lane], c);
        #pragma unroll
        for (int v = 0; v < 8; ++v)
            Y[(v + 8 * khalf) * 64 + nt * 16 + col] = (_Float16)gelu_f(c[v]);
    }
    lds_fence();

    a0 = ldA(Y, col, khalf, 0);
    a1 = ldA(Y, col, khalf, 1);

    int nb = nvec[b];
    const v16h* W2 = (const v16h*)w2f;
    for (int nt = 0; nt < 4; ++nt) {
        float bb = db2[nt * 16 + col];
        v8f c = {bb, bb, bb, bb, bb, bb, bb, bb};
        c = wmma_f16(a0, W2[(nt * 2 + 0) * 32 + lane], c);
        c = wmma_f16(a1, W2[(nt * 2 + 1) * 32 + lane], c);
        float* xcol = outX + (size_t)b * 1024 + nt * 16 + col;
        #pragma unroll
        for (int v = 0; v < 8; ++v) {
            int p = v + 8 * khalf;
            xcol[p * 64] = (p < nb) ? c[v] : 0.0f;
        }
    }
    if (lane < 16) {
        int p = lane;
        outMask[(size_t)b * 16 + p]  = (p < nb) ? 1.0f : 0.0f;
        outBatch[(size_t)b * 16 + p] = (float)b;
    }
}

extern "C" void kernel_launch(void* const* d_in, const int* in_sizes, int n_in,
                              void* d_out, int out_size, void* d_ws, size_t ws_size,
                              hipStream_t stream) {
    const float* z      = (const float*)d_in[0];
    const float* sp_w1  = (const float*)d_in[1];
    const float* sp_b1  = (const float*)d_in[2];
    const float* sp_w2  = (const float*)d_in[3];
    const float* sp_b2  = (const float*)d_in[4];
    const float* pe_w1  = (const float*)d_in[5];
    const float* pe_b1  = (const float*)d_in[6];
    const float* pe_w2  = (const float*)d_in[7];
    const float* pe_b2  = (const float*)d_in[8];
    const float* dec_w1 = (const float*)d_in[9];
    const float* dec_b1 = (const float*)d_in[10];
    const float* dec_w2 = (const float*)d_in[11];
    const float* dec_b2 = (const float*)d_in[12];

    int B = in_sizes[0] / 64;

    char* ws = (char*)d_ws;
    float*     ws_pe  = (float*)(ws + 0);          // 16*64 f32      = 4096 B
    _Float16*  f_w1   = (_Float16*)(ws + 4096);    // 4*2*512 halfs  = 8192 B
    _Float16*  f_w2   = (_Float16*)(ws + 12288);   //                = 8192 B
    _Float16*  f_sw1  = (_Float16*)(ws + 20480);   // 3*2*512 halfs  = 6144 B
    _Float16*  f_sw2  = (_Float16*)(ws + 26624);   // 1*2*512 halfs  = 2048 B
    int*       nvec   = (int*)(ws + 28672);        // B ints

    float* outX     = (float*)d_out;
    float* outMask  = outX + (size_t)B * 1024;
    float* outBatch = outMask + (size_t)B * 16;
    float* outNpred = outBatch + (size_t)B * 16;

    k0_pack<<<1, 256, 0, stream>>>(sp_w1, sp_w2, pe_w1, pe_b1, pe_w2, pe_b2,
                                   dec_w1, dec_w2, ws_pe, f_w1, f_w2, f_sw1, f_sw2);

    int tiles = (B + 15) / 16;
    int g1 = (tiles + WAVES1 - 1) / WAVES1;
    k1_sp<<<g1, 128, 0, stream>>>(z, sp_b1, sp_b2, f_sw1, f_sw2, nvec, outNpred, B);

    int g2 = (B + WAVES2 - 1) / WAVES2;
    k2_dec<<<g2, 256, 0, stream>>>(z, dec_b1, dec_b2, ws_pe, f_w1, f_w2, nvec,
                                   outX, outMask, outBatch, B);
}
